// VectorQuantizer_36541581754908
// MI455X (gfx1250) — matrix operand order repaired and checked by reading, not compiled, not run on hardware
//
#include <hip/hip_runtime.h>

// ---------------------------------------------------------------------------
// VectorQuantizer for MI455X (gfx1250, wave32, WMMA)
// x:[8,4096,256] f32, codebook:[8192,256] f32 -> out = codebook[argmin dist]
// bf16 WMMA GEMM (scores = ||c||^2 - 2 x.c), per-lane running argmin,
// shfl-xor final reduce, fp32 gather. Codebook chunks staged to LDS via
// GLOBAL_LOAD_ASYNC_TO_LDS (ASYNCcnt double-buffer) when available.
// ---------------------------------------------------------------------------

typedef unsigned int u32;
typedef unsigned short u16;
typedef __attribute__((ext_vector_type(2)))  u32   u32x2;
typedef __attribute__((ext_vector_type(4)))  u32   u32x4;
typedef __attribute__((ext_vector_type(4)))  int   i32x4;
typedef __attribute__((ext_vector_type(4)))  float f32x4;
typedef __attribute__((ext_vector_type(16))) __bf16 v16bf;
typedef __attribute__((ext_vector_type(8)))  float  v8f;

#define VQ_TOKENS (8 * 4096)   // 32768
#define VQ_D      256
#define VQ_K      8192
#define VQ_CPC    32           // codes per chunk (two 16-wide N tiles)
#define VQ_NCH    (VQ_K / VQ_CPC)  // 256 chunks
#define VQ_TPB    128          // tokens per block (8 waves x 16)

// ---- async copy to LDS (gfx1250) -----------------------------------------
#if __has_builtin(__builtin_amdgcn_global_load_async_to_lds_b128)
#define VQ_ASYNC 1
#else
#define VQ_ASYNC 0
#endif

#if __has_builtin(__builtin_amdgcn_s_wait_asynccnt)
#define VQ_WAIT_ASYNC(nn) __builtin_amdgcn_s_wait_asynccnt(nn)
#else
#define VQ_WAIT_ASYNC(nn) asm volatile("s_wait_asynccnt %0" ::"n"(nn))
#endif

__device__ __forceinline__ u16 f32_to_bf16_rne(float f) {
    u32 u = __float_as_uint(f);
    u32 r = u + 0x7FFFu + ((u >> 16) & 1u);
    return (u16)(r >> 16);
}

// Reinterpret two 16B chunks as a 16-element bf16 vector (all-trivial union).
union BfPack {
    u32x4 q[2];
    v16bf v;
};

// ---------------------------------------------------------------------------
// Prep 1: x fp32 -> bf16 (each thread converts 4 elements)
// ---------------------------------------------------------------------------
__global__ __launch_bounds__(256) void vq_x_prep(const float* __restrict__ x,
                                                 u16* __restrict__ xb) {
    const int t = blockIdx.x * 256 + threadIdx.x;  // 0 .. TOKENS*D/4-1
    f32x4 v = *(const f32x4*)(x + (size_t)t * 4);
    u32 lo = (u32)f32_to_bf16_rne(v.x) | ((u32)f32_to_bf16_rne(v.y) << 16);
    u32 hi = (u32)f32_to_bf16_rne(v.z) | ((u32)f32_to_bf16_rne(v.w) << 16);
    u32x2 o = {lo, hi};
    *(u32x2*)(xb + (size_t)t * 4) = o;
}

// ---------------------------------------------------------------------------
// Prep 2: codebook fp32 -> bf16 + per-code squared norm. One block per code.
// ---------------------------------------------------------------------------
__global__ __launch_bounds__(256) void vq_cb_prep(const float* __restrict__ cb,
                                                  u16* __restrict__ cbb,
                                                  float* __restrict__ cnorm) {
    const int k   = blockIdx.x;   // code index
    const int tid = threadIdx.x;  // 0..255 == d
    float v = cb[(size_t)k * VQ_D + tid];
    cbb[(size_t)k * VQ_D + tid] = f32_to_bf16_rne(v);
    float s = v * v;
    #pragma unroll
    for (int m = 16; m >= 1; m >>= 1) s += __shfl_xor(s, m, 32);
    __shared__ float wsum[8];
    if ((tid & 31) == 0) wsum[tid >> 5] = s;
    __syncthreads();
    if (tid < 8) {
        float t2 = wsum[tid];
        #pragma unroll
        for (int m = 4; m >= 1; m >>= 1) t2 += __shfl_xor(t2, m, 32);
        if (tid == 0) cnorm[k] = t2;
    }
}

// ---------------------------------------------------------------------------
// Main: WMMA score GEMM + running argmin.
// Block = 256 threads = 8 waves; wave w owns tokens [m0, m0+16) (A tile in
// VGPRs). Each chunk = 32 codes (16 KB bf16 in LDS) -> two independent
// 8-deep WMMA chains reusing the A registers.
// ---------------------------------------------------------------------------
__global__ __launch_bounds__(256) void vq_argmin(const u16* __restrict__ xb,
                                                 const u16* __restrict__ cbb,
                                                 const float* __restrict__ cnorm,
                                                 int* __restrict__ out_idx) {
#if VQ_ASYNC
    __shared__ __align__(16) u16 ldsB[2][VQ_CPC * VQ_D];  // 2 x 16 KB
#else
    __shared__ __align__(16) u16 ldsB[1][VQ_CPC * VQ_D];  // 16 KB
#endif

    const int tid  = threadIdx.x;
    const int lane = tid & 31;
    const int wave = tid >> 5;
    const int n    = lane & 15;  // B column / code-within-tile, C/D column
    const int hi   = lane >> 4;  // lane-half selector (K split)
    const int m0   = blockIdx.x * VQ_TPB + wave * 16;

    // ---- Load the 16x256 bf16 A tile into VGPRs (16-bit A layout) ----------
    // lanes 0-15 hold K = {kc*32 + 0..7, kc*32 + 16..23}; lanes 16-31 hold +8.
    v16bf A[8];
    {
        const u16* p = xb + (size_t)(m0 + n) * VQ_D + hi * 8;
        #pragma unroll
        for (int kc = 0; kc < 8; ++kc) {
            BfPack u;
            u.q[0] = *(const u32x4*)(p + kc * 32);
            u.q[1] = *(const u32x4*)(p + kc * 32 + 16);
            A[kc] = u.v;
        }
    }

    float best[8];
    int   bidx[8];
    #pragma unroll
    for (int i = 0; i < 8; ++i) { best[i] = 3.0e38f; bidx[i] = 0; }

#if VQ_ASYNC
    // Each thread async-copies its 64B slice of a chunk straight into LDS.
    // Builtin params are (int4 __device__*, int4 __shared__*, imm, imm);
    // HIP's generic pointers convert implicitly once the pointee type matches.
    auto issue_stage = [&](int chunk, int buf) {
        i32x4* g = (i32x4*)(cbb + (size_t)chunk * (VQ_CPC * VQ_D) + tid * 32);
        i32x4* l = (i32x4*)(&ldsB[buf][tid * 32]);
        #pragma unroll
        for (int q = 0; q < 4; ++q) {
            __builtin_amdgcn_global_load_async_to_lds_b128(g + q, l + q, 0, 0);
        }
    };
    issue_stage(0, 0);
#else
    // Register double-buffer: prefetch chunk 0 staging slice (64B per thread).
    u32x4 r[4];
    {
        const u32x4* src = (const u32x4*)(cbb) + (size_t)tid * 4;
        #pragma unroll
        for (int q = 0; q < 4; ++q) r[q] = src[q];
    }
#endif

    for (int c = 0; c < VQ_NCH; ++c) {
#if VQ_ASYNC
        const int buf = c & 1;
        if (c + 1 < VQ_NCH) {
            issue_stage(c + 1, buf ^ 1);  // overlap copy with compute
            VQ_WAIT_ASYNC(4);             // chunk c's 4 loads done (in-order)
        } else {
            VQ_WAIT_ASYNC(0);
        }
        __syncthreads();  // all waves' slices of chunk c visible
#else
        const int buf = 0;
        __syncthreads();  // everyone done reading chunk c-1
        #pragma unroll
        for (int q = 0; q < 4; ++q) *(u32x4*)&ldsB[0][tid * 32 + q * 8] = r[q];
        if (c + 1 < VQ_NCH) {
            const u32x4* src =
                (const u32x4*)(cbb + (size_t)(c + 1) * (VQ_CPC * VQ_D)) +
                (size_t)tid * 4;
            #pragma unroll
            for (int q = 0; q < 4; ++q) r[q] = src[q];
        }
        __syncthreads();  // chunk c staged
#endif
        if (c + 2 < VQ_NCH) {
            // gfx1250 global_prefetch_b8 of the chunk after next
            __builtin_prefetch(cbb + (size_t)(c + 2) * (VQ_CPC * VQ_D) + tid * 32,
                               0, 1);
        }

        const float cn0 = cnorm[c * VQ_CPC + n];
        const float cn1 = cnorm[c * VQ_CPC + 16 + n];

        // Two independent accumulator chains (tile0: codes n, tile1: codes 16+n)
        v8f acc0 = {0.f, 0.f, 0.f, 0.f, 0.f, 0.f, 0.f, 0.f};
        v8f acc1 = {0.f, 0.f, 0.f, 0.f, 0.f, 0.f, 0.f, 0.f};
        #pragma unroll
        for (int kc = 0; kc < 8; ++kc) {
            // B layout (32x16 bf16): lanes 0-15 K=kc*32+{0..7,16..23}, lanes 16-31 +8.
            BfPack b0, b1;
            const u16* bp0 = &ldsB[buf][n * VQ_D + kc * 32 + hi * 8];
            const u16* bp1 = &ldsB[buf][(16 + n) * VQ_D + kc * 32 + hi * 8];
            b0.q[0] = *(const u32x4*)(bp0);
            b0.q[1] = *(const u32x4*)(bp0 + 16);
            b1.q[0] = *(const u32x4*)(bp1);
            b1.q[1] = *(const u32x4*)(bp1 + 16);
            acc0 = __builtin_amdgcn_wmma_f32_16x16x32_bf16(
                false, A[kc], false, b0.v, (short)0, acc0, false, false);
            acc1 = __builtin_amdgcn_wmma_f32_16x16x32_bf16(
                false, A[kc], false, b1.v, (short)0, acc1, false, false);
        }

        const int code0 = c * VQ_CPC + n;
        const int code1 = code0 + 16;
        #pragma unroll
        for (int i = 0; i < 8; ++i) {
            float s0 = fmaf(-2.0f, acc0[i], cn0);  // ||c||^2 - 2 x.c
            if (s0 < best[i]) { best[i] = s0; bidx[i] = code0; }
            float s1 = fmaf(-2.0f, acc1[i], cn1);
            if (s1 < best[i]) { best[i] = s1; bidx[i] = code1; }
        }

#if VQ_ASYNC
        __syncthreads();  // done reading buf before it gets re-targeted
#endif
    }

    // ---- Final argmin across the 16 lanes holding each token row -----------
    // C/D layout: VGPR i, lanes 0-15 -> token m0+i ; lanes 16-31 -> m0+8+i.
    #pragma unroll
    for (int m = 1; m < 16; m <<= 1) {
        #pragma unroll
        for (int i = 0; i < 8; ++i) {
            float ob = __shfl_xor(best[i], m, 32);
            int   oi = __shfl_xor(bidx[i], m, 32);
            if (ob < best[i] || (ob == best[i] && oi < bidx[i])) {
                best[i] = ob;
                bidx[i] = oi;
            }
        }
    }
    if (n == 0) {  // lanes 0 and 16
        #pragma unroll
        for (int i = 0; i < 8; ++i) out_idx[m0 + hi * 8 + i] = bidx[i];
    }
}

// ---------------------------------------------------------------------------
// Gather: out[token,:] = codebook[idx[token],:] in fp32 (16B vectors)
// ---------------------------------------------------------------------------
__global__ __launch_bounds__(256) void vq_gather(const float* __restrict__ cb,
                                                 const int* __restrict__ idx,
                                                 float* __restrict__ out) {
    const int t     = blockIdx.x * 256 + threadIdx.x;  // 0 .. TOKENS*64-1
    const int token = t >> 6;                          // 64 f32x4 per token
    const int c4    = t & 63;
    const int k     = idx[token];
    ((u32x4*)out)[(size_t)token * 64 + c4] =
        ((const u32x4*)cb)[(size_t)k * 64 + c4];
}

// ---------------------------------------------------------------------------
extern "C" void kernel_launch(void* const* d_in, const int* in_sizes, int n_in,
                              void* d_out, int out_size, void* d_ws, size_t ws_size,
                              hipStream_t stream) {
    const float* x  = (const float*)d_in[0];  // [8,4096,256] f32
    const float* cb = (const float*)d_in[1];  // [8192,256]   f32
    float* out      = (float*)d_out;          // [8,4096,256] f32

    char* ws = (char*)d_ws;
    // workspace layout:
    //   xb    : TOKENS*D  bf16  = 16 MB
    //   cbb   : K*D       bf16  =  4 MB
    //   cnorm : K         f32   = 32 KB
    //   idx   : TOKENS    i32   = 128 KB
    u16*   xb    = (u16*)(ws);
    u16*   cbb   = (u16*)(ws + (size_t)VQ_TOKENS * VQ_D * 2);
    float* cnorm = (float*)(ws + (size_t)VQ_TOKENS * VQ_D * 2 + (size_t)VQ_K * VQ_D * 2);
    int*   idx   = (int*)(ws + (size_t)VQ_TOKENS * VQ_D * 2 + (size_t)VQ_K * VQ_D * 2 +
                          (size_t)VQ_K * 4);

    // 1) x -> bf16
    vq_x_prep<<<(VQ_TOKENS * VQ_D) / (4 * 256), 256, 0, stream>>>(x, xb);
    // 2) codebook -> bf16 + ||c||^2
    vq_cb_prep<<<VQ_K, 256, 0, stream>>>(cb, cbb, cnorm);
    // 3) WMMA scores + argmin (256 blocks x 8 waves, 16 tokens/wave)
    vq_argmin<<<VQ_TOKENS / VQ_TPB, 256, 0, stream>>>(xb, cbb, cnorm, idx);
    // 4) gather fp32 codebook rows
    vq_gather<<<(VQ_TOKENS * (VQ_D / 4)) / 256, 256, 0, stream>>>(cb, idx, out);
}
